// HistoryUnit_45165876085136
// MI455X (gfx1250) — matrix operand order repaired and checked by reading, not compiled, not run on hardware
//
#include <hip/hip_runtime.h>

// ---------------------------------------------------------------------------
// MI455X (gfx1250) HistoryUnit forward.
//
// Dataflow pruning: only h1[:16]/h2[:16] are consumed and the S6 scan is
// chunk-local (chunk==16), so each FABIS6 block needs just 19 input positions
// (16 + conv halo 3) and 16 output positions.
//
// Layout: activations are batch-major [b][s][d] with a 4-position zero halo
// below each sequence, so
//   * every conv is ONE WMMA GEMM with K = ksz*D (im2col via contiguity),
//   * conv zero-padding is plain pointer offset into the halo (no predication
//     in the K-loop -> EXEC stays full, pure load+wmma inner loop),
//   * snip reshape (B,16*256) is free.
// All GEMMs: bf16 operands, f32 WMMA accumulation; LN/activations in f32.
// S6 step = dual-operand GEMM (x@A^T + h@B^T) with fused tanh*sel epilogue
// (double-buffered H), then h@C^T with fused +x*Dv epilogue.
// ---------------------------------------------------------------------------

typedef __bf16 bf16_t;
typedef __attribute__((ext_vector_type(16))) __bf16 v16bf;
typedef __attribute__((ext_vector_type(8)))  __bf16 v8bf;
typedef __attribute__((ext_vector_type(8)))  float  v8f;

#define ACT_NONE    0
#define ACT_RELU    1
#define ACT_SIG     2
#define ACT_TANHMUL 3   // val = tanh(val) * aux[xrow]
#define ACT_ADDXDV  4   // val += aux[xrow] * Dv[n]

static constexpr int B_   = 32;        // batch
static constexpr int D1   = 512;       // d_model
static constexpr int D2   = 1024;      // 2*d_model (block2 dim)
static constexpr int HT   = 16;        // history tokens == chunk
static constexpr int HALO = 4;         // max conv pad
static constexpr int SPAD = 19;        // live input positions per block
static constexpr int SP   = HALO + SPAD;   // 23 physical rows per batch in x buffers
static constexpr int MO   = HT * B_;   // 512 live output rows

// ---------------------------------------------------------------------------
// Dual-operand NT WMMA GEMM with affine row maps.
//   acc = sum_k A1[a1row(r),k]*W1[n,k]  (+ sum_k A2[a2row(r),k]*W2[n,k])
//   row maps: a1row = a1Off + tm*a1GS + lsub*a1Sc   (likewise a2 / c / aux)
//   C[crow, colOff+n] = act(acc + bias1[n] + bias2[n])
// One wave = one 16x16 tile, 4 waves/block (32x32 macro-tile).
// No load predication: halo/padded buffers guarantee validity; only the
// final store is masked on n<N.
// ---------------------------------------------------------------------------
__global__ __launch_bounds__(128)
void gemm2_nt_wmma(const bf16_t* __restrict__ A1, const bf16_t* __restrict__ W1,
                   int K1, int lda1,
                   const bf16_t* __restrict__ A2, const bf16_t* __restrict__ W2,
                   int K2, int lda2,
                   const float* __restrict__ bias1, const float* __restrict__ bias2,
                   float* __restrict__ C, bf16_t* __restrict__ Cbf,
                   const float* __restrict__ aux, const float* __restrict__ Dv,
                   int M, int N, int ldc,
                   int a1Off, int a1GS, int a1Sc,
                   int a2Off, int a2GS, int a2Sc,
                   int cOff,  int cGS,  int cSc,
                   int xOff,  int xGS,  int xSc,
                   int colOff, int act)
{
    const int wave = threadIdx.x >> 5;
    const int lane = threadIdx.x & 31;
    const int half = lane >> 4;
    const int lsub = lane & 15;

    const int tm = blockIdx.x * 2 + (wave >> 1);
    const int tn = blockIdx.y * 2 + (wave & 1);
    if (tm * 16 >= M || tn * 16 >= N) return;          // wave-uniform

    const int n = tn * 16 + lsub;                      // B-fragment lane = column

    // ---- operand 1 ----
    {
        const int a1row = a1Off + tm * a1GS + lsub * a1Sc;
        const bf16_t* Ar = A1 + (size_t)a1row * lda1;
        const bf16_t* Wr = W1 + (size_t)n * K1;
        v8f acc = {};
        for (int k0 = 0; k0 < K1; k0 += 32) {
            v16bf af;
            {   // A 16x32 layout: half 0 -> K {0..15}, half 1 -> K {16..31}
                v8bf lo = *(const v8bf*)(Ar + k0 + half * 16);
                v8bf hi = *(const v8bf*)(Ar + k0 + 8 + half * 16);
                #pragma unroll
                for (int i = 0; i < 8; ++i) { af[i] = lo[i]; af[8 + i] = hi[i]; }
            }
            // B 32x16 layout: lanes 0-15 hold K=k0..k0+15, lanes 16-31 K=k0+16..k0+31
            v16bf bfv = *(const v16bf*)(Wr + k0 + half * 16);
            acc = __builtin_amdgcn_wmma_f32_16x16x32_bf16(false, af, false, bfv,
                                                          (short)0, acc, false, false);
        }
        if (A2) {
            const int a2row = a2Off + tm * a2GS + lsub * a2Sc;
            const bf16_t* Ar2 = A2 + (size_t)a2row * lda2;
            const bf16_t* Wr2 = W2 + (size_t)n * K2;
            for (int k0 = 0; k0 < K2; k0 += 32) {
                v16bf af;
                v8bf lo = *(const v8bf*)(Ar2 + k0 + half * 16);
                v8bf hi = *(const v8bf*)(Ar2 + k0 + 8 + half * 16);
                #pragma unroll
                for (int i = 0; i < 8; ++i) { af[i] = lo[i]; af[8 + i] = hi[i]; }
                v16bf bfv = *(const v16bf*)(Wr2 + k0 + half * 16);
                acc = __builtin_amdgcn_wmma_f32_16x16x32_bf16(false, af, false, bfv,
                                                              (short)0, acc, false, false);
            }
        }

        if (n >= N) return;                            // store mask only
        const float bv = (bias1 ? bias1[n] : 0.0f) + (bias2 ? bias2[n] : 0.0f);
        #pragma unroll
        for (int v = 0; v < 8; ++v) {
            const int m    = half * 8 + v;             // C layout: VGPR v -> M = v + half*8
            const int crow = cOff + tm * cGS + m * cSc;
            const size_t idx = (size_t)crow * ldc + colOff + n;
            float val = acc[v] + bv;
            if (act == ACT_RELU) {
                val = val > 0.0f ? val : 0.0f;
            } else if (act == ACT_SIG) {
                val = 1.0f / (1.0f + __expf(-val));
            } else if (act == ACT_TANHMUL) {
                const int xr = xOff + tm * xGS + m * xSc;
                val = tanhf(val) * aux[(size_t)xr * ldc + colOff + n];
            } else if (act == ACT_ADDXDV) {
                const int xr = xOff + tm * xGS + m * xSc;
                val += aux[(size_t)xr * ldc + colOff + n] * Dv[n];
            }
            if (C)   C[idx]   = val;
            if (Cbf) Cbf[idx] = (bf16_t)val;
        }
    }
}

// ---------------------------------------------------------------------------
// LayerNorm over last dim of X (+X2), one 256-thread block per row.
// Output row map: orow = oOff + (row/16)*oGS + (row%16)*oSc  (handles halo
// placement and the [b][s] -> [s][b] permute for the final output).
// ---------------------------------------------------------------------------
__global__ __launch_bounds__(256)
void ln_kernel(const float* __restrict__ X, const float* __restrict__ X2,
               const float* __restrict__ g, const float* __restrict__ bta,
               float* __restrict__ Y, bf16_t* __restrict__ Ybf,
               int D, int ldx, int ldy, int ycol,
               int oOff, int oGS, int oSc)
{
    const int row = blockIdx.x;
    const float* x  = X  + (size_t)row * ldx;
    const float* x2 = X2 ? X2 + (size_t)row * ldx : nullptr;
    __shared__ float red[256];

    float s = 0.0f;
    for (int d = threadIdx.x; d < D; d += 256) {
        float v = x[d] + (x2 ? x2[d] : 0.0f); s += v;
    }
    red[threadIdx.x] = s; __syncthreads();
    for (int o = 128; o > 0; o >>= 1) {
        if (threadIdx.x < o) red[threadIdx.x] += red[threadIdx.x + o];
        __syncthreads();
    }
    const float mean = red[0] / D; __syncthreads();

    float vs = 0.0f;
    for (int d = threadIdx.x; d < D; d += 256) {
        float v = x[d] + (x2 ? x2[d] : 0.0f) - mean; vs += v * v;
    }
    red[threadIdx.x] = vs; __syncthreads();
    for (int o = 128; o > 0; o >>= 1) {
        if (threadIdx.x < o) red[threadIdx.x] += red[threadIdx.x + o];
        __syncthreads();
    }
    const float rstd = rsqrtf(red[0] / D + 1e-5f); __syncthreads();

    const int orow = oOff + (row / 16) * oGS + (row % 16) * oSc;
    for (int d = threadIdx.x; d < D; d += 256) {
        float v = (x[d] + (x2 ? x2[d] : 0.0f) - mean) * rstd * g[d] + bta[d];
        if (Y)   Y  [(size_t)orow * ldy + ycol + d] = v;
        if (Ybf) Ybf[(size_t)orow * ldy + ycol + d] = (bf16_t)v;
    }
}

// ------------------------------- elementwise -------------------------------
// rows 0..Nsrc-1 copied (f32->bf16), rows Nsrc..Npad-1 zero-filled
__global__ void cvt_rows(const float* __restrict__ s, bf16_t* __restrict__ d,
                         int Nsrc, int K, int n /*Npad*K*/)
{
    int i = blockIdx.x * 256 + threadIdx.x;
    if (i >= n) return;
    d[i] = (i < Nsrc * K) ? (bf16_t)s[i] : (bf16_t)0.0f;
}

// conv weight (O, D, ksz) -> padded dense (Opad, ksz*D): wt[o][k*D+i] = w[o][i][k]
__global__ void cvt_conv(const float* __restrict__ w, bf16_t* __restrict__ d,
                         int O, int D, int ksz, int n /*Opad*ksz*D*/)
{
    int i = blockIdx.x * 256 + threadIdx.x;
    if (i >= n) return;
    int KD = ksz * D;
    int o = i / KD, r = i % KD;
    int k = r / D,  dd = r % D;
    d[i] = (o < O) ? (bf16_t)w[((size_t)o * D + dd) * ksz + k] : (bf16_t)0.0f;
}

__global__ void fill_zero_bf16(bf16_t* __restrict__ p, int n)
{ int i = blockIdx.x * 256 + threadIdx.x; if (i < n) p[i] = (bf16_t)0.0f; }

// x1 (batch-major, halo): rows b*SP+q; q<4 zero; q-4<16 token; else long_x+PE
__global__ void build_x1_kernel(const float* __restrict__ long_x,
                                const float* __restrict__ htok,
                                bf16_t* __restrict__ x1, int n)
{
    int i = blockIdx.x * 256 + threadIdx.x;
    if (i >= n) return;
    int d  = i % D1;
    int pr = i / D1;
    int b  = pr / SP;
    int q  = pr % SP;
    float v = 0.0f;
    if (q >= HALO) {
        int s = q - HALO;
        if (s < HT) {
            v = htok[s * D1 + d];                      // (16,1,D) broadcast over batch
        } else {
            int p  = s - HT;                           // long_x position 0..2
            int dd = d & ~1;
            float den = __expf(-(float)dd * (9.210340372f / (float)D1)); // ln(1e4)/D
            float a   = (float)p * den;
            float pe  = (d & 1) ? cosf(a) : sinf(a);
            v = long_x[((size_t)p * B_ + b) * D1 + d] + pe;
        }
    }
    x1[i] = (bf16_t)v;
}

// ---------------------------------------------------------------------------
// Host orchestration
// ---------------------------------------------------------------------------
static inline char* bump(char*& p, size_t bytes) {
    char* r = p;
    p += (bytes + 255) & ~(size_t)255;
    return r;
}

extern "C" void kernel_launch(void* const* d_in, const int* in_sizes, int n_in,
                              void* d_out, int out_size, void* d_ws, size_t ws_size,
                              hipStream_t stream)
{
    (void)in_sizes; (void)n_in; (void)out_size; (void)ws_size;
    auto F = [&](int i) { return (const float*)d_in[i]; };
    char* wp = (char*)d_ws;

    struct Map { int off, gs, sc; };
    const Map STD{0, 16, 1};                       // identity: row = tm*16 + m

    auto gemm = [&](const bf16_t* A1, const bf16_t* W1, int K1, int lda1,
                    const bf16_t* A2, const bf16_t* W2, int K2, int lda2,
                    const float* b1, const float* b2,
                    float* C, bf16_t* Cbf, const float* aux, const float* Dv,
                    int M, int N, int ldc, Map a1, Map a2, Map c, Map x,
                    int colOff, int act) {
        dim3 g((M + 31) / 32, (N + 31) / 32);
        gemm2_nt_wmma<<<g, 128, 0, stream>>>(A1, W1, K1, lda1, A2, W2, K2, lda2,
                                             b1, b2, C, Cbf, aux, Dv, M, N, ldc,
                                             a1.off, a1.gs, a1.sc, a2.off, a2.gs, a2.sc,
                                             c.off, c.gs, c.sc, x.off, x.gs, x.sc,
                                             colOff, act);
    };
    auto cvt = [&](int idx, int Nsrc, int K, int Npad) -> bf16_t* {
        int n = Npad * K;
        bf16_t* d = (bf16_t*)bump(wp, (size_t)n * sizeof(bf16_t));
        cvt_rows<<<(n + 255) / 256, 256, 0, stream>>>(F(idx), d, Nsrc, K, n);
        return d;
    };
    auto cvtc = [&](int idx, int O, int Opad, int D, int ksz) -> bf16_t* {
        int n = Opad * ksz * D;
        bf16_t* d = (bf16_t*)bump(wp, (size_t)n * sizeof(bf16_t));
        cvt_conv<<<(n + 255) / 256, 256, 0, stream>>>(F(idx), d, O, D, ksz, n);
        return d;
    };

    // -------- parameter indices (flattened setup_inputs() dict order) --------
    struct CSpec { int widx, bidx, O, Opad, ksz, pad, colOff; };
    const CSpec cs1[6] = {{3,4,170,176,2,1,0},{5,6,170,176,3,1,170},{7,8,172,176,4,2,340},
                          {9,10,170,176,2,1,0},{11,12,170,176,4,2,170},{13,14,172,176,8,4,340}};
    const CSpec cs2[6] = {{39,40,341,352,2,1,0},{41,42,341,352,3,1,341},{43,44,342,352,4,2,682},
                          {45,46,341,352,2,1,0},{47,48,341,352,4,2,341},{49,50,342,352,8,4,682}};
    const int b1ln[6] = {15,16,17,18,19,20};   // tfa g/b, cfa g/b, feat g/b
    const int b2ln[6] = {51,52,53,54,55,56};
    const int s6f1 = 21, s6b1 = 29, fin1g = 37, fin1b = 38;
    const int s6f2 = 57, s6b2 = 65, fin2g = 73, fin2b = 74;
    const int iEncW=75,iEncB=76,iHistW=77,iHistB=78,iHtokW=79,iHtokB=80;
    const int iSnHW=81,iSnHB=82,iW1=83,iB1=84,iW2=85,iB2=86,iN2g=87,iN2b=88;

    // -------- weights -> bf16 --------
    bf16_t *cw1[6], *cw2[6];
    for (int c = 0; c < 6; ++c) cw1[c] = cvtc(cs1[c].widx, cs1[c].O, cs1[c].Opad, D1, cs1[c].ksz);
    for (int c = 0; c < 6; ++c) cw2[c] = cvtc(cs2[c].widx, cs2[c].O, cs2[c].Opad, D2, cs2[c].ksz);

    struct S6W { bf16_t *A, *Bm, *Cm, *sW; const float *Dv, *sb, *lg, *lb; };
    auto mk_s6 = [&](int base, int D) {
        S6W s;
        s.A  = cvt(base + 0, D, D, D); s.Bm = cvt(base + 1, D, D, D);
        s.Cm = cvt(base + 2, D, D, D); s.sW = cvt(base + 4, D, D, D);
        s.Dv = F(base + 3); s.sb = F(base + 5); s.lg = F(base + 6); s.lb = F(base + 7);
        return s;
    };
    S6W s6_1f = mk_s6(s6f1, D1), s6_1b = mk_s6(s6b1, D1);
    S6W s6_2f = mk_s6(s6f2, D2), s6_2b = mk_s6(s6b2, D2);

    bf16_t* encW  = cvt(iEncW,  D2, D1, D2);
    bf16_t* histW = cvt(iHistW, D1, 2 * D2, D1);
    bf16_t* htokW = cvt(iHtokW, D1, D2, D1);
    bf16_t* snHW  = cvt(iSnHW,  256, D2, 256);
    bf16_t* W1bf  = cvt(iW1,    1024, 4096, 1024);
    bf16_t* W2bf  = cvt(iW2,    200, 1024, 208);       // pad 200 -> 208 rows

    // -------- activation scratch --------
    bf16_t* x1bf = (bf16_t*)bump(wp, (size_t)B_ * SP * D1 * 2);   // halo'd block1 input
    bf16_t* x2bf = (bf16_t*)bump(wp, (size_t)B_ * SP * D2 * 2);   // halo'd block2 input
    bf16_t* hfbf = (bf16_t*)bump(wp, (size_t)MO * 2 * D2 * 2);    // block2 out (b,s,2048)
    auto fb = [&](size_t n) { return (float*)bump(wp, n * sizeof(float)); };
    auto bb = [&](size_t n) { return (bf16_t*)bump(wp, n * sizeof(bf16_t)); };

    // FABIS6 over the 16 live positions; input xbf is halo'd batch-major.
    auto run_block = [&](int D, const CSpec* cs, bf16_t* const* cw,
                         const int* lnI, const S6W& sf, const S6W& sb,
                         int finG, int finB, const bf16_t* xbf,
                         bf16_t* outbf, Map outMap) {
        float*  tfa  = fb((size_t)MO * D);
        float*  cfa  = fb((size_t)MO * D);
        float*  agg  = fb((size_t)MO * D);
        bf16_t* agbf = bb((size_t)MO * D);
        float*  selb = fb((size_t)MO * D);
        bf16_t* Ha   = bb((size_t)B_ * D);
        bf16_t* Hb   = bb((size_t)B_ * D);
        float*  Ybuf = fb((size_t)MO * D);
        float*  cat  = fb((size_t)MO * 2 * D);

        // 6 convs, one fused-K GEMM each (K = ksz*D, zero-pad via halo)
        for (int c = 0; c < 6; ++c) {
            float* dst = (c < 3) ? tfa : cfa;
            const CSpec& s = cs[c];
            gemm(xbf, cw[c], s.ksz * D, D, nullptr, nullptr, 0, 0,
                 F(s.bidx), nullptr, dst, nullptr, nullptr, nullptr,
                 MO, s.O, D, Map{HALO - s.pad, SP, 1}, STD, STD, STD,
                 s.colOff, ACT_NONE);
        }
        ln_kernel<<<MO, 256, 0, stream>>>(tfa, nullptr, F(lnI[0]), F(lnI[1]), tfa, nullptr, D, D, D, 0, 0, 16, 1);
        ln_kernel<<<MO, 256, 0, stream>>>(cfa, nullptr, F(lnI[2]), F(lnI[3]), cfa, nullptr, D, D, D, 0, 0, 16, 1);
        ln_kernel<<<MO, 256, 0, stream>>>(tfa, cfa,     F(lnI[4]), F(lnI[5]), agg, agbf,   D, D, D, 0, 0, 16, 1);

        // bidirectional chunk-local S6 (rows: b*16+s; step p touches row b*16+p)
        for (int dir = 0; dir < 2; ++dir) {
            const S6W& w = dir ? sb : sf;
            gemm(agbf, w.sW, D, D, nullptr, nullptr, 0, 0, w.sb, nullptr,
                 selb, nullptr, nullptr, nullptr, MO, D, D, STD, STD, STD, STD, 0, ACT_SIG);
            fill_zero_bf16<<<(B_ * D + 255) / 256, 256, 0, stream>>>(Ha, B_ * D);
            bf16_t* Hin = Ha; bf16_t* Hout = Hb;
            for (int t = 0; t < HT; ++t) {
                int p = dir ? (HT - 1 - t) : t;
                Map gat{p, 256, 16};                  // row b -> b*16+p
                // H' = tanh(x_p@A^T + H@B^T) * sel_p     (dual GEMM, fused epilogue)
                gemm(agbf, w.A, D, D, Hin, w.Bm, D, D, nullptr, nullptr,
                     nullptr, Hout, selb, nullptr, B_, D, D,
                     gat, STD, STD, gat, 0, ACT_TANHMUL);
                // Y_p = H'@C^T + x_p*Dv                  (fused epilogue, scatter rows)
                gemm(Hout, w.Cm, D, D, nullptr, nullptr, 0, 0, nullptr, nullptr,
                     Ybuf, nullptr, agg, w.Dv, B_, D, D,
                     STD, STD, gat, gat, 0, ACT_ADDXDV);
                bf16_t* tmp = Hin; Hin = Hout; Hout = tmp;
            }
            ln_kernel<<<MO, 256, 0, stream>>>(Ybuf, nullptr, w.lg, w.lb, cat, nullptr,
                                              D, D, 2 * D, dir ? D : 0, 0, 16, 1);
        }
        ln_kernel<<<MO, 256, 0, stream>>>(cat, nullptr, F(finG), F(finB), nullptr, outbf,
                                          2 * D, 2 * D, 2 * D, 0, outMap.off, outMap.gs, outMap.sc);
    };

    // -------- forward pass --------
    build_x1_kernel<<<(B_ * SP * D1 + 255) / 256, 256, 0, stream>>>(F(0), F(2), x1bf, B_ * SP * D1);
    fill_zero_bf16<<<(B_ * SP * D2 + 255) / 256, 256, 0, stream>>>(x2bf, B_ * SP * D2);

    // block1 -> htok_enc straight into x2bf halo'd rows (b*SP + 4 + s)
    run_block(D1, cs1, cw1, b1ln, s6_1f, s6_1b, fin1g, fin1b, x1bf, x2bf, Map{HALO, SP, 1});

    // enc_proj of the 3 live encoded positions -> x2bf rows (b*SP + 20 + e)
    bf16_t* encbf = cvt(1, 3 * B_, D1, 3 * B_);        // encoded_x[0:3] in [e][b] order
    for (int e = 0; e < 3; ++e)
        gemm(encbf, encW, D1, D1, nullptr, nullptr, 0, 0, F(iEncB), nullptr,
             nullptr, x2bf, nullptr, nullptr, B_, D2, D2,
             Map{e * B_, 16, 1}, STD, Map{HALO + HT + e, 16 * SP, SP}, STD, 0, ACT_NONE);

    // block2 -> hist_final (b,s,2048)
    run_block(D2, cs2, cw2, b2ln, s6_2f, s6_2b, fin2g, fin2b, x2bf, hfbf, STD);

    // hist_enc = LN( hist_final@histW^T + b + htok_enc@htokW^T + b2 )  (dual GEMM)
    Map xmap{HALO, SP, 1};                             // htok_enc rows inside x2bf
    float* heF = fb((size_t)MO * D1);
    gemm(hfbf, histW, 2 * D2, 2 * D2, x2bf, htokW, D2, D2, F(iHistB), F(iHtokB),
         heF, nullptr, nullptr, nullptr, MO, D1, D1, STD, xmap, STD, STD, 0, ACT_NONE);
    // norm2 with [b][s] -> [s][b] permute into d_out
    ln_kernel<<<MO, 256, 0, stream>>>(heF, nullptr, F(iN2g), F(iN2b),
                                      (float*)d_out, nullptr, D1, D1, D1, 0, 0, 1, B_);

    // snip head: batch-major layout makes the (B,16*256) reshape free
    bf16_t* snip2 = bb((size_t)B_ * 4096);
    bf16_t* snip3 = bb((size_t)B_ * 1024);
    gemm(x2bf, snHW, D2, D2, nullptr, nullptr, 0, 0, F(iSnHB), nullptr,
         nullptr, snip2, nullptr, nullptr, MO, 256, 256, xmap, STD, STD, STD, 0, ACT_RELU);
    gemm(snip2, W1bf, 4096, 4096, nullptr, nullptr, 0, 0, F(iB1), nullptr,
         nullptr, snip3, nullptr, nullptr, B_, 1024, 1024, STD, STD, STD, STD, 0, ACT_RELU);
    gemm(snip3, W2bf, 1024, 1024, nullptr, nullptr, 0, 0, F(iB2), nullptr,
         (float*)d_out + HT * B_ * D1, nullptr, nullptr, nullptr,
         B_, 200, 200, STD, STD, STD, STD, 0, ACT_NONE);
}
